// TiledMemristorLinear_67585605370373
// MI455X (gfx1250) — matrix operand order repaired and checked by reading, not compiled, not run on hardware
//
#include <hip/hip_runtime.h>
#include <stdint.h>

typedef __attribute__((ext_vector_type(16))) _Float16 v16h;
typedef __attribute__((ext_vector_type(8)))  _Float16 v8h;
typedef __attribute__((ext_vector_type(8)))  float    v8f;

#define R_PITCH 132   // dwords per LDS row of an r tile (128 data + 4 pad, TDM pad descriptor)

// LDS layout (dynamic):
//   s_rb0 : [128][132] f32  r tile, channel buffer 0 (TDM destination)   67584 B
//   s_rb1 : [128][132] f32  r tile, channel buffer 1 (TDM destination)   67584 B
//   s_low : [2][16][128] f32 signed low-conductance                      16384 B
//   s_dh  : [2][16][128] f16 signed (high-low), WMMA A operand            8192 B
//   s_kfh : [16][128] f16    sqrt(noise factor)*256                       4096 B
//   s_S   : [16] f32         per-row Σ_m low (both channels)                64 B
#define OFF_RB0  0
#define OFF_RB1  67584
#define OFF_LOW  135168
#define OFF_DH   151552
#define OFF_KFH  159744
#define OFF_S    163840
#define SMEM_BYTES (OFF_S + 64)

#if __has_builtin(__builtin_amdgcn_tensor_load_to_lds) && __has_builtin(__builtin_amdgcn_s_wait_tensorcnt)
#define HAVE_TDM 1
#else
#define HAVE_TDM 0
#endif

#if HAVE_TDM
// Issue one TDM DMA: 128x128 f32 tile, global row-major -> LDS with 4-dword pad
// every 128 dwords (=> 132-dword LDS pitch). Wave-level op; EXEC ignored.
static __device__ __forceinline__ void tdm_issue(const float* g, void* lds_ptr) {
  typedef unsigned int u32x4 __attribute__((ext_vector_type(4)));
  typedef int          i32x4 __attribute__((ext_vector_type(4)));
  typedef int          i32x8 __attribute__((ext_vector_type(8)));
  const unsigned long long ga = (unsigned long long)(uintptr_t)g;
  u32x4 g0;
  g0[0] = 1u;                                   // D# group0: count=1, user descriptor
  g0[1] = (unsigned int)(uintptr_t)lds_ptr;     // lds_addr (byte offset, addr[31:0])
  g0[2] = (unsigned int)ga;                     // global_addr[31:0]
  g0[3] = (unsigned int)((ga >> 32) & 0x01FFFFFFull) | 0x80000000u; // addr[56:32] | type=2
  i32x8 g1;
  g1[0] = (int)((2u << 16) | (1u << 20) | (6u << 22) | (3u << 25)); // 4B elems, pad_en,
                                                                    // interval=128dw, amount=4dw
  g1[1] = (int)(128u << 16);   // tensor_dim0 = 128   (bits 63:48)
  g1[2] = (int)(128u << 16);   // tensor_dim1 = 128   (bits 95:80)
  g1[3] = (int)(128u << 16);   // tile_dim0   = 128   (bits 127:112)
  g1[4] = (int)128u;           // tile_dim1   = 128, tile_dim2 = 0
  g1[5] = (int)128u;           // tensor_dim0_stride = 128
  g1[6] = (int)(16384u << 16); // tensor_dim1_stride[15:0] = 16384 (unused for 2D tile)
  g1[7] = 0;
  i32x4 z4 = {0, 0, 0, 0};
#if defined(__clang_major__) && (__clang_major__ >= 23)
  i32x8 z8 = {0, 0, 0, 0, 0, 0, 0, 0};
  __builtin_amdgcn_tensor_load_to_lds(g0, g1, z4, z4, z8, 0);
#else
  __builtin_amdgcn_tensor_load_to_lds(g0, g1, z4, z4, 0);
#endif
}
#endif

// grid = (64 row-tiles, TK=2, 4 (i,j) planes); block = 256 = 8 wave32.
// Writes one (i,j) plane of bit-weighted ADC partials into ws[z][1024][256].
// Requires ws_size >= 4*1024*256*4 = 4 MiB.
__global__ __launch_bounds__(256) void memristor_partial(
    const float* __restrict__ x, const float* __restrict__ pl,
    const float* __restrict__ ph, const float* __restrict__ r,
    float* __restrict__ ws)
{
  extern __shared__ char smem[];
  float*    s_rb0 = (float*)(smem + OFF_RB0);
  float*    s_rb1 = (float*)(smem + OFF_RB1);
  float*    s_low = (float*)(smem + OFF_LOW);
  _Float16* s_dh  = (_Float16*)(smem + OFF_DH);
  _Float16* s_kfh = (_Float16*)(smem + OFF_KFH);
  float*    s_S   = (float*)(smem + OFF_S);

  const int tid  = threadIdx.x;
  const int lane = tid & 31;
  const int wave = tid >> 5;
  const int kblk = blockIdx.y;
  const int btb  = blockIdx.x * 16;
  const int zij  = blockIdx.z;                // (i,j) plane
  const int i    = zij >> 1;
  const int j    = zij & 1;
  const int ij   = zij * 2 + kblk;
  const int oloc = wave * 16 + (lane & 15);
  const int koff = (lane >> 4) * 8;
  const float wgt = (i == 0) ? 2.0f : 1.0f;   // 2^bit, bit = 1-i

  const float* rt0 = r + (size_t)(ij * 2 + 0) * 16384;
  const float* rt1 = r + (size_t)(ij * 2 + 1) * 16384;

#if HAVE_TDM
  if (wave == 0) tdm_issue(rt0, smem + OFF_RB0);   // overlap DMA with phase 1
#else
  __builtin_prefetch(rt0 + tid * 64, 0, 1);        // global_prefetch_b8
#endif

  // ---- phase 1: DAC quantize + polynomial conductances (both channels) ----
  {
    float cl0[7], cl1[7], ch0[6], ch1[6];
    const float* pcl0 = pl + (size_t)(ij * 2 + 0) * 7;
    const float* pcl1 = pl + (size_t)(ij * 2 + 1) * 7;
    const float* pch0 = ph + (size_t)(ij * 2 + 0) * 6;
    const float* pch1 = ph + (size_t)(ij * 2 + 1) * 6;
    #pragma unroll
    for (int p = 0; p < 7; ++p) { cl0[p] = pcl0[p]; cl1[p] = pcl1[p]; }
    #pragma unroll
    for (int p = 0; p < 6; ++p) { ch0[p] = pch0[p]; ch1[p] = pch1[p]; }

    const int t  = tid >> 4;
    const int m0 = (tid & 15) * 8;
    const float* xrow = x + (size_t)(btb + t) * 256 + j * 128;
    #pragma unroll
    for (int q = 0; q < 8; ++q) {
      const int m = m0 + q;
      float qv = rintf(xrow[m] * 128.0f);
      qv = fminf(fmaxf(qv, -128.0f), 128.0f);
      const float v  = qv * (0.6f / 128.0f);
      const float av = fabsf(v);
      // Kf = 4*kB*T*BW/(|v|+eps) + 2*e*BW ; store sqrt(Kf)*256 as f16
      const float kf = 1.6567788e-28f / (av + 1e-12f) + 5.4365637e-8f;
      s_kfh[t * 128 + m] = (_Float16)(sqrtf(kf) * 256.0f);
      float lo0 = cl0[6], lo1 = cl1[6];
      float hi0 = ch0[5], hi1 = ch1[5];
      #pragma unroll
      for (int p = 5; p >= 0; --p) { lo0 = fmaf(lo0, v, cl0[p]); lo1 = fmaf(lo1, v, cl1[p]); }
      #pragma unroll
      for (int p = 4; p >= 0; --p) { hi0 = fmaf(hi0, v, ch0[p]); hi1 = fmaf(hi1, v, ch1[p]); }
      s_low[(0 * 16 + t) * 128 + m] =  lo0;                 // channel sign folded
      s_dh [(0 * 16 + t) * 128 + m] = (_Float16)(hi0 - lo0);
      s_low[(1 * 16 + t) * 128 + m] = -lo1;
      s_dh [(1 * 16 + t) * 128 + m] = (_Float16)(lo1 - hi1);
    }
  }
  __syncthreads();
  if (tid < 16) {                         // per-row deterministic Σ_m low (both c)
    float s = 0.0f;
    #pragma unroll 4
    for (int m = 0; m < 128; ++m)
      s += s_low[tid * 128 + m] + s_low[(16 + tid) * 128 + m];
    s_S[tid] = s;
  }

  v8f pacc = {};
  for (int c = 0; c < 2; ++c) {
    float* scur = (c == 0) ? s_rb0 : s_rb1;
#if HAVE_TDM
    if (wave == 0) {
      if (c == 0) {
        tdm_issue(rt1, smem + OFF_RB1);              // prefetch next tile via TDM
        __builtin_amdgcn_s_wait_tensorcnt(1);        // tile 0 landed, tile 1 in flight
      } else {
        __builtin_amdgcn_s_wait_tensorcnt(0);        // tile 1 landed
      }
    }
#else
    {
      const float* rt = (c == 0) ? rt0 : rt1;
      #pragma unroll
      for (int q = 0; q < 16; ++q) {
        const int f4 = q * 256 + tid;                // coalesced b128 reads
        const float4 vv = *(const float4*)(rt + (size_t)f4 * 4);
        const int m = f4 >> 5;
        const int o = (f4 & 31) * 4;
        *(float4*)&scur[m * R_PITCH + o] = vv;       // aligned: pitch, o multiples of 4
      }
    }
#endif
    __syncthreads();

    // ---- deterministic crossbar GEMM: pacc += diff_c (16x128) * r_c (128x16) ----
    const _Float16* ab   = s_dh + (c * 16 + (lane & 15)) * 128 + koff;
    const float*    bcol = scur + (lane & 15);       // N column of r tile
    #pragma unroll
    for (int kk = 0; kk < 4; ++kk) {
      const v8h a0 = *(const v8h*)(ab + kk * 32);        // K = 32kk+koff .. +7
      const v8h a1 = *(const v8h*)(ab + kk * 32 + 16);   // K = 32kk+koff+16 .. +23
      const v16h A = __builtin_shufflevector(a0, a1, 0,1,2,3,4,5,6,7,8,9,10,11,12,13,14,15);
      const int kbase = kk * 32 + koff;
      v16h Bm;
      #pragma unroll
      for (int hh = 0; hh < 16; ++hh)
        Bm[hh] = (_Float16)bcol[(kbase + (hh & 7) + ((hh >> 3) << 4)) * R_PITCH];
      pacc = __builtin_amdgcn_wmma_f32_16x16x32_f16(
          false, A, false, Bm, (short)0, pacc, false, false);
    }

    // ---- stochastic path: pacc[u] += Σ_m n(t,m,o) * sqrt(|raw|*Kf) ----
    const uint32_t seed = (0x9E3779B9u * (uint32_t)(i * 4 + j * 2 + kblk + 1)) ^
                          (0x7F4A7C15u * (uint32_t)(c + 1));
    const float* rp = scur + oloc;
    #pragma unroll
    for (int u = 0; u < 8; ++u) {
      const int t = u + koff;                            // matches C-tile M mapping
      const uint32_t tt = (uint32_t)((btb + t) & 255);   // time index (noise b-invariant)
      uint32_t hm = (seed ^ (tt * 0x01000193u)) + (uint32_t)oloc * 0x27D4EB2Fu;
      const _Float16* dp = s_dh  + (c * 16 + t) * 128;
      const float*    lp = s_low + (c * 16 + t) * 128;
      const _Float16* kp = s_kfh + t * 128;
      float accN = 0.0f;
      #pragma unroll 4
      for (int m = 0; m < 128; ++m) {
        const float rr  = rp[m * R_PITCH];
        const float dd  = (float)dp[m];
        const float raw = fmaf(dd, rr, lp[m]);
        const float sig = sqrtf(fabsf(raw)) * (float)kp[m];   // 256*sigma
        hm += 0x85EBCA6Bu;
        uint32_t h = hm ^ (hm >> 15);
        h *= 0x2C1B3C6Du;
        h ^= h >> 13;
        // uniform -> zero-mean unit-variance, 1/256 sigma rescale folded in
        const float n = fmaf((float)h, 3.1502e-12f, -6.7658233e-3f);
        accN = fmaf(n, sig, accN);
      }
      pacc[u] += accN;
    }
  } // c

  // ---- ADC fake-quant + bit weighting, write partial plane ----
  float* wsp = ws + (size_t)zij * (1024 * 256);
  const int og = kblk * 128 + oloc;
  #pragma unroll
  for (int u = 0; u < 8; ++u) {
    const int t = u + koff;
    float q = rintf((pacc[u] + s_S[t]) * 2053120.0f);   // I_SCALE * 256
    q = fminf(fmaxf(q, -2048.0f), 2048.0f);
    wsp[(size_t)(btb + t) * 256 + og] = wgt * q * 0.00390625f;
  }
}

// Deterministic reduction of the 4 (i,j) planes + bias.
__global__ __launch_bounds__(256) void memristor_reduce(
    const float* __restrict__ ws, const float* __restrict__ bias,
    float* __restrict__ out)
{
  const int idx = blockIdx.x * 256 + threadIdx.x;
  const float b = bias[idx & 255];
  float s = ws[idx];
  s += ws[idx + 1 * 1024 * 256];
  s += ws[idx + 2 * 1024 * 256];
  s += ws[idx + 3 * 1024 * 256];
  out[idx] = s + b;
}

extern "C" void kernel_launch(void* const* d_in, const int* in_sizes, int n_in,
                              void* d_out, int out_size, void* d_ws, size_t ws_size,
                              hipStream_t stream) {
  (void)in_sizes; (void)n_in; (void)out_size; (void)ws_size;
  const float* x  = (const float*)d_in[0];
  const float* pl = (const float*)d_in[1];
  const float* ph = (const float*)d_in[2];
  const float* r  = (const float*)d_in[3];
  const float* bs = (const float*)d_in[4];
  float* out = (float*)d_out;
  float* ws  = (float*)d_ws;     // needs 4 MiB: 4 planes x 1024 x 256 f32

  dim3 grid(64, 2, 4);           // row-tiles x TK x (i,j) planes
  dim3 block(256, 1, 1);         // 8 wave32
  hipLaunchKernelGGL(memristor_partial, grid, block, SMEM_BYTES, stream,
                     x, pl, ph, r, ws);
  hipLaunchKernelGGL(memristor_reduce, dim3(1024, 1, 1), block, 0, stream,
                     ws, bs, out);
}
